// MultiHeadAttention_31877247271608
// MI455X (gfx1250) — matrix operand order repaired and checked by reading, not compiled, not run on hardware
//
#include <hip/hip_runtime.h>
#include <hip/hip_bf16.h>

// MHA: B=4, S=2048, D=1024, H=16, Hd=64.  bf16 WMMA pipeline, fp32 accumulate.
// Workspace layout (bf16 elements): xb | Wq | Wk | Wv | Wo | Q | K | Vt | ctx
// total = 5*B*S*D + 4*D*D bf16 = ~92.3 MB.

typedef __attribute__((ext_vector_type(16))) __bf16 v16bf;
typedef __attribute__((ext_vector_type(8)))  __bf16 v8bf;
typedef __attribute__((ext_vector_type(8)))  float  v8f;

constexpr int B_ = 4, S_ = 2048, D_ = 1024, H_ = 16, HD_ = 64;
constexpr size_t NX = (size_t)B_ * S_ * D_;   // 8388608 elements
constexpr size_t NW = (size_t)D_ * D_;        // 1048576 elements

__device__ __forceinline__ v8f wmma_bf16(v16bf a, v16bf b, v8f c) {
  return __builtin_amdgcn_wmma_f32_16x16x32_bf16(
      /*neg_a=*/false, a, /*neg_b=*/false, b,
      /*c_mod=*/(short)0, c, /*reuse_a=*/false, /*reuse_b=*/false);
}

// A-fragment: 16x32 bf16 tile, row-major with `stride` elements between rows.
// Lane L holds row (L&15); lanes 0-15 carry K={0..7,16..23}, lanes 16-31 K={8..15,24..31}.
__device__ __forceinline__ v16bf load_frag_a(const __bf16* base, int stride, int lane) {
  const int row = lane & 15;
  const int k0  = (lane >> 4) ? 8 : 0;
  const __bf16* p = base + row * stride + k0;
  v8bf lo = *(const v8bf*)(p);        // K = k0 .. k0+7
  v8bf hi = *(const v8bf*)(p + 16);   // K = k0+16 .. k0+23
  return __builtin_shufflevector(lo, hi, 0,1,2,3,4,5,6,7,8,9,10,11,12,13,14,15);
}

// B-fragment: 32x16 bf16 tile where COLUMN n lives contiguously at base + n*stride.
// Lanes 0-15: col=lane, K={0..7,16..23} ; lanes 16-31: col=lane-16, K={8..15,24..31}.
__device__ __forceinline__ v16bf load_frag_b(const __bf16* base, int stride, int lane) {
  const int n  = lane & 15;
  const int k0 = (lane >> 4) ? 8 : 0;
  const __bf16* p = base + n * stride + k0;
  v8bf lo = *(const v8bf*)(p);        // K = k0 .. k0+7
  v8bf hi = *(const v8bf*)(p + 16);   // K = k0+16 .. k0+23
  return __builtin_shufflevector(lo, hi, 0,1,2,3,4,5,6,7,8,9,10,11,12,13,14,15);
}

// Async global->LDS copy of 16 bytes per lane (tracked by ASYNCcnt).
// VDST = per-lane LDS byte offset, VADDR = per-lane 64-bit global address.
__device__ __forceinline__ void async_ld16(unsigned lds_off, const __bf16* gaddr) {
  asm volatile("global_load_async_to_lds_b128 %0, %1, off"
               :: "v"(lds_off), "v"(gaddr)
               : "memory");
}
__device__ __forceinline__ void wait_async0() {
  asm volatile("s_wait_asynccnt 0x0" ::: "memory");
}
__device__ __forceinline__ unsigned lds_off32(const void* p) {
  // LDS aperture keeps the AS3 byte offset in addr[31:0].
  return (unsigned)(size_t)p;
}

// ---------------- fp32 -> bf16 conversion (8 elements / thread) ----------------
__global__ __launch_bounds__(256) void cvt_bf16_kernel(const float* __restrict__ in,
                                                       __bf16* __restrict__ out,
                                                       int n8) {
  int i = blockIdx.x * blockDim.x + threadIdx.x;
  if (i >= n8) return;
  const float4* p = (const float4*)in + (size_t)i * 2;
  float4 a = p[0], b = p[1];
  v8bf o;
  o[0] = (__bf16)a.x; o[1] = (__bf16)a.y; o[2] = (__bf16)a.z; o[3] = (__bf16)a.w;
  o[4] = (__bf16)b.x; o[5] = (__bf16)b.y; o[6] = (__bf16)b.z; o[7] = (__bf16)b.w;
  *((v8bf*)out + i) = o;
}

// ---------------- fused QKV projection ----------------
// y[s,e] = sum_d x[s,d] * W[e,d] + bias[e]
// Wave tile 64x64; block = 8 waves -> 128(M) x 256(N); grid.z selects Q/K/V.
// Q scaled by 1/sqrt(Hd) and stored [B,H,S,Hd]; K stored [B,H,S,Hd]; V stored [B,H,Hd,S].
__global__ __launch_bounds__(256) void qkv_proj_kernel(
    const __bf16* __restrict__ xb,
    const __bf16* __restrict__ wq, const __bf16* __restrict__ wk, const __bf16* __restrict__ wv,
    const float* __restrict__ bq, const float* __restrict__ bk, const float* __restrict__ bv,
    __bf16* __restrict__ Qb, __bf16* __restrict__ Kb, __bf16* __restrict__ Vtb) {
  const int lane = threadIdx.x & 31;
  const int wav  = threadIdx.x >> 5;
  const int m0 = blockIdx.x * 128 + (wav >> 2) * 64;
  const int n0 = blockIdx.y * 256 + (wav & 3) * 64;
  const int mode = blockIdx.z;
  const __bf16* W   = (mode == 0) ? wq : (mode == 1) ? wk : wv;
  const float* bias = (mode == 0) ? bq : (mode == 1) ? bk : bv;

  v8f acc[4][4];
  #pragma unroll
  for (int i = 0; i < 4; ++i)
    #pragma unroll
    for (int j = 0; j < 4; ++j) acc[i][j] = (v8f)(0.0f);

  for (int k = 0; k < D_; k += 32) {
    v16bf a[4], bf[4];
    #pragma unroll
    for (int i = 0; i < 4; ++i)
      a[i] = load_frag_a(xb + (size_t)(m0 + i * 16) * D_ + k, D_, lane);
    #pragma unroll
    for (int j = 0; j < 4; ++j)
      bf[j] = load_frag_b(W + (size_t)(n0 + j * 16) * D_ + k, D_, lane);
    #pragma unroll
    for (int i = 0; i < 4; ++i)
      #pragma unroll
      for (int j = 0; j < 4; ++j)
        acc[i][j] = wmma_bf16(a[i], bf[j], acc[i][j]);
  }

  const int hlf = lane >> 4, col = lane & 15;
  #pragma unroll
  for (int j = 0; j < 4; ++j) {
    const int e  = n0 + j * 16 + col;
    const float be = bias[e];
    const int h = e >> 6, hd = e & 63;
    #pragma unroll
    for (int i = 0; i < 4; ++i) {
      #pragma unroll
      for (int r = 0; r < 8; ++r) {
        const int m = m0 + i * 16 + r + hlf * 8;     // global row in [0, B*S)
        const int b = m >> 11, s = m & (S_ - 1);
        const float val = acc[i][j][r] + be;
        if (mode == 0) {
          Qb[((size_t)(b * H_ + h) * S_ + s) * HD_ + hd] = (__bf16)(val * 0.125f);
        } else if (mode == 1) {
          Kb[((size_t)(b * H_ + h) * S_ + s) * HD_ + hd] = (__bf16)val;
        } else {
          Vtb[((size_t)(b * H_ + h) * HD_ + hd) * S_ + s] = (__bf16)val;
        }
      }
    }
  }
}

// ---------------- flash attention ----------------
// Block = 8 waves, all on one (b,h), q rows [qblk, qblk+128).  Keys stream in
// chunks of 32 through double-buffered LDS tiles filled with async global->LDS
// copies (ASYNCcnt), overlapped with the WMMA + online-softmax pipeline.
__global__ __launch_bounds__(256) void flash_attn_kernel(
    const __bf16* __restrict__ Qb, const __bf16* __restrict__ Kb,
    const __bf16* __restrict__ Vtb, __bf16* __restrict__ ctx) {
  __shared__ __align__(16) __bf16 Kt[2][32 * 64];    // 2 x 4 KB key tiles   [key][feat]
  __shared__ __align__(16) __bf16 Vt[2][64 * 32];    // 2 x 4 KB value tiles [feat][key]
  __shared__ __align__(16) __bf16 Pbuf[8][16 * 32];  // per-wave P staging (8 KB)

  const int tid  = threadIdx.x;
  const int lane = tid & 31;
  const int wav  = tid >> 5;
  const int hlf  = lane >> 4, col = lane & 15;
  const int bh    = blockIdx.y;                      // b*H + h
  const int qbase = (blockIdx.x * 8 + wav) * 16;

  const __bf16* Qp = Qb  + (size_t)bh * S_ * HD_;
  const __bf16* Kp = Kb  + (size_t)bh * S_ * HD_;
  const __bf16* Vp = Vtb + (size_t)bh * HD_ * S_;

  // Cooperative async staging: K tile is 32 rows x 64 feats = 4 KB contiguous;
  // V tile is 64 feat-rows x 32 keys (row stride S_ in global, 32 in LDS).
  const int vf = tid >> 2, vq = tid & 3;             // V: 4 x 16B loads per feat row
  auto stage = [&](int kb, int buf) {
    async_ld16(lds_off32(&Kt[buf][tid * 8]), Kp + (size_t)kb * HD_ + tid * 8);
    async_ld16(lds_off32(&Vt[buf][vf * 32 + vq * 8]),
               Vp + (size_t)vf * S_ + kb + vq * 8);
  };

  // Q fragments (16 x 64), loaded once (Q already carries the 1/sqrt(Hd) scale)
  const v16bf aq0 = load_frag_a(Qp + (size_t)qbase * HD_,      HD_, lane);
  const v16bf aq1 = load_frag_a(Qp + (size_t)qbase * HD_ + 32, HD_, lane);

  v8f O[4];
  #pragma unroll
  for (int f = 0; f < 4; ++f) O[f] = (v8f)(0.0f);
  float m_i[8], l_i[8];
  #pragma unroll
  for (int r = 0; r < 8; ++r) { m_i[r] = -1e30f; l_i[r] = 0.0f; }

  __bf16* pb = &Pbuf[wav][0];

  stage(0, 0);                                       // prologue: tile 0 in flight

  for (int kb = 0; kb < S_; kb += 32) {
    const int cur = (kb >> 5) & 1;
    wait_async0();                                   // own async copies done
    __syncthreads();                                 // whole block's copies visible
    if (kb + 32 < S_) stage(kb + 32, 1 - cur);       // overlap next tile with compute

    // ---- S = Q * K^T  (16 x 32 tile across two accumulators) ----
    const __bf16* KL = &Kt[cur][0];
    v8f s0 = (v8f)(0.0f), s1 = (v8f)(0.0f);
    {
      v16bf k00 = load_frag_b(KL,                 HD_, lane);
      v16bf k01 = load_frag_b(KL + 32,            HD_, lane);
      v16bf k10 = load_frag_b(KL + 16 * HD_,      HD_, lane);
      v16bf k11 = load_frag_b(KL + 16 * HD_ + 32, HD_, lane);
      s0 = wmma_bf16(aq0, k00, s0);
      s0 = wmma_bf16(aq1, k01, s0);
      s1 = wmma_bf16(aq0, k10, s1);
      s1 = wmma_bf16(aq1, k11, s1);
    }

    // ---- online softmax (row = r + hlf*8; 16-lane half-wave reductions) ----
    float p0[8], p1[8], corr[8];
    #pragma unroll
    for (int r = 0; r < 8; ++r) {
      float t = fmaxf(s0[r], s1[r]);
      t = fmaxf(t, __shfl_xor(t, 1));
      t = fmaxf(t, __shfl_xor(t, 2));
      t = fmaxf(t, __shfl_xor(t, 4));
      t = fmaxf(t, __shfl_xor(t, 8));
      const float mn = fmaxf(m_i[r], t);
      corr[r] = __expf(m_i[r] - mn);
      p0[r]   = __expf(s0[r] - mn);
      p1[r]   = __expf(s1[r] - mn);
      float u = p0[r] + p1[r];
      u += __shfl_xor(u, 1);
      u += __shfl_xor(u, 2);
      u += __shfl_xor(u, 4);
      u += __shfl_xor(u, 8);
      l_i[r] = l_i[r] * corr[r] + u;
      m_i[r] = mn;
    }
    #pragma unroll
    for (int f = 0; f < 4; ++f)
      #pragma unroll
      for (int r = 0; r < 8; ++r) O[f][r] *= corr[r];

    // ---- P through LDS: accumulator layout -> A-fragment layout ----
    #pragma unroll
    for (int r = 0; r < 8; ++r) {
      pb[(r + hlf * 8) * 32 + col]      = (__bf16)p0[r];
      pb[(r + hlf * 8) * 32 + 16 + col] = (__bf16)p1[r];
    }
    const v16bf pf = load_frag_a(pb, 32, lane);      // ds_load_b128 x2

    // ---- O += P * V  (LDS V tile [feat][key]: columns contiguous) ----
    #pragma unroll
    for (int f = 0; f < 4; ++f) {
      v16bf vfr = load_frag_b(&Vt[cur][f * 16 * 32], 32, lane);
      O[f] = wmma_bf16(pf, vfr, O[f]);
    }
  }

  // ---- epilogue: normalize and write ctx[B,S,D] (bf16) ----
  float linv[8];
  #pragma unroll
  for (int r = 0; r < 8; ++r) linv[r] = 1.0f / l_i[r];
  const int b = bh >> 4, h = bh & 15;
  #pragma unroll
  for (int f = 0; f < 4; ++f) {
    #pragma unroll
    for (int r = 0; r < 8; ++r) {
      const int s = qbase + r + hlf * 8;
      const int e = h * HD_ + f * 16 + col;
      ctx[((size_t)b * S_ + s) * D_ + e] = (__bf16)(O[f][r] * linv[r]);
    }
  }
}

// ---------------- output projection: out = ctx @ Wo^T + bo (fp32 out) ----------------
__global__ __launch_bounds__(256) void out_proj_kernel(
    const __bf16* __restrict__ ctx, const __bf16* __restrict__ wo,
    const float* __restrict__ bo, float* __restrict__ out) {
  const int lane = threadIdx.x & 31;
  const int wav  = threadIdx.x >> 5;
  const int m0 = blockIdx.x * 128 + (wav >> 2) * 64;
  const int n0 = blockIdx.y * 256 + (wav & 3) * 64;

  v8f acc[4][4];
  #pragma unroll
  for (int i = 0; i < 4; ++i)
    #pragma unroll
    for (int j = 0; j < 4; ++j) acc[i][j] = (v8f)(0.0f);

  for (int k = 0; k < D_; k += 32) {
    v16bf a[4], bf[4];
    #pragma unroll
    for (int i = 0; i < 4; ++i)
      a[i] = load_frag_a(ctx + (size_t)(m0 + i * 16) * D_ + k, D_, lane);
    #pragma unroll
    for (int j = 0; j < 4; ++j)
      bf[j] = load_frag_b(wo + (size_t)(n0 + j * 16) * D_ + k, D_, lane);
    #pragma unroll
    for (int i = 0; i < 4; ++i)
      #pragma unroll
      for (int j = 0; j < 4; ++j)
        acc[i][j] = wmma_bf16(a[i], bf[j], acc[i][j]);
  }

  const int hlf = lane >> 4, col = lane & 15;
  #pragma unroll
  for (int j = 0; j < 4; ++j) {
    const int e = n0 + j * 16 + col;
    const float be = bo[e];
    #pragma unroll
    for (int i = 0; i < 4; ++i) {
      #pragma unroll
      for (int r = 0; r < 8; ++r) {
        const int m = m0 + i * 16 + r + hlf * 8;
        out[(size_t)m * D_ + e] = acc[i][j][r] + be;
      }
    }
  }
}

extern "C" void kernel_launch(void* const* d_in, const int* in_sizes, int n_in,
                              void* d_out, int out_size, void* d_ws, size_t ws_size,
                              hipStream_t stream) {
  const float* x  = (const float*)d_in[0];
  const float* Wq = (const float*)d_in[1];
  const float* bq = (const float*)d_in[2];
  const float* Wk = (const float*)d_in[3];
  const float* bk = (const float*)d_in[4];
  const float* Wv = (const float*)d_in[5];
  const float* bv = (const float*)d_in[6];
  const float* Wo = (const float*)d_in[7];
  const float* bo = (const float*)d_in[8];
  float* out = (float*)d_out;

  __bf16* ws = (__bf16*)d_ws;
  __bf16* xb  = ws;
  __bf16* wqb = xb  + NX;
  __bf16* wkb = wqb + NW;
  __bf16* wvb = wkb + NW;
  __bf16* wob = wvb + NW;
  __bf16* Qb  = wob + NW;
  __bf16* Kb  = Qb  + NX;
  __bf16* Vtb = Kb  + NX;
  __bf16* ctx = Vtb + NX;

  // 1) fp32 -> bf16 conversions
  {
    int n8x = (int)(NX / 8), n8w = (int)(NW / 8);
    cvt_bf16_kernel<<<(n8x + 255) / 256, 256, 0, stream>>>(x,  xb,  n8x);
    cvt_bf16_kernel<<<(n8w + 255) / 256, 256, 0, stream>>>(Wq, wqb, n8w);
    cvt_bf16_kernel<<<(n8w + 255) / 256, 256, 0, stream>>>(Wk, wkb, n8w);
    cvt_bf16_kernel<<<(n8w + 255) / 256, 256, 0, stream>>>(Wv, wvb, n8w);
    cvt_bf16_kernel<<<(n8w + 255) / 256, 256, 0, stream>>>(Wo, wob, n8w);
  }

  // 2) fused QKV projection: grid = (M/128, N/256, 3)
  qkv_proj_kernel<<<dim3((B_ * S_) / 128, D_ / 256, 3), 256, 0, stream>>>(
      xb, wqb, wkb, wvb, bq, bk, bv, Qb, Kb, Vtb);

  // 3) flash attention: grid = (S/128 q-tiles, B*H)
  flash_attn_kernel<<<dim3(S_ / 128, B_ * H_), 256, 0, stream>>>(Qb, Kb, Vtb, ctx);

  // 4) output projection: grid = (M/128, N/256)
  out_proj_kernel<<<dim3((B_ * S_) / 128, D_ / 256), 256, 0, stream>>>(ctx, wob, bo, out);
}
